// SparsePoolingLayer_22196390986277
// MI455X (gfx1250) — compile-verified, matrix operand order checked by reading, not run on hardware
//
#include <hip/hip_runtime.h>

typedef __attribute__((ext_vector_type(16))) _Float16 v16h;
typedef __attribute__((ext_vector_type(8)))  float    v8f;

#define TAU 0.2f
#define EPS 0.01f

// Problem dims
#define NB     16
#define CIN    64
#define HIN    64
#define WIN    64
#define COUT   256
#define HOUT   60
#define WOUT   60
#define PIXIMG (HOUT*WOUT)          // 3600
#define NPIX   (NB*PIXIMG)          // 57600
#define KFF    (CIN*25)             // 1600
#define NELEM  (NPIX*COUT)          // 14,745,600

// Workspace layout (bytes)
#define FLAG_OFF 0
#define NUM_OFF  8
#define DEN_OFF  12
#define BFF_OFF  1024
#define BR_OFF   (1u<<20)
#define AF_OFF   (2u<<20)                                 // f16 a, frag layout
#define U0_OFF   (AF_OFF + (size_t)NELEM*2)
#define U_OFF    (U0_OFF + (size_t)NELEM*4)

#define ITER_BLOCKS 450
#define TILES_PER_BLOCK 4          // 450*4 = 1800 tiles of 32 pixels = 57600
#define NITER 48

// ---------------------------------------------------------------------------
// Pack an [O=256][K] row-major fp32 matrix into f16 WMMA B-fragments:
// fragment (kt, ot) occupies 512 contiguous halves; lane L holds 16 halves:
//   j=0..15 -> k = kt*32 + (L/16)*8 + (j&7) + (j>>3)*16,  o = ot*16 + (L&15)
// ---------------------------------------------------------------------------
__global__ void pack_b_kernel(const float* __restrict__ src,
                              _Float16* __restrict__ dst, int K) {
    int tid = blockIdx.x * blockDim.x + threadIdx.x;
    if (tid >= K * 256) return;
    int frag = tid >> 9;
    int r    = tid & 511;
    int L    = r >> 4;
    int j    = r & 15;
    int ot   = frag & 15;
    int kt   = frag >> 4;
    int k = kt*32 + (L>>4)*8 + (j&7) + ((j>>3)<<4);
    int o = ot*16 + (L&15);
    dst[tid] = (_Float16)src[o*K + k];
}

// ---------------------------------------------------------------------------
// u0 = conv5x5 VALID via implicit GEMM with v_wmma_f32_16x16x32_f16.
// Block = 256 threads (8 wave32s), 32 pixels x 256 out-channels per block.
// ---------------------------------------------------------------------------
__global__ __launch_bounds__(256)
void conv_wmma_kernel(const float* __restrict__ inp,
                      const v16h*  __restrict__ Bff,
                      float*       __restrict__ u0) {
    __shared__ _Float16 As[1024];          // 32 pixels x 32 k, fragment layout
    const int t    = threadIdx.x;
    const int lane = t & 31;
    const int wave = t >> 5;
    const int mt   = wave >> 2;
    const int og   = wave & 3;
    const int mp0  = blockIdx.x * 32;

    v8f acc[4] = {};

    for (int k0 = 0; k0 < KFF; k0 += 32) {
        __syncthreads();
#pragma unroll
        for (int hh = 0; hh < 4; ++hh) {    // im2col gather -> frag layout
            int h   = t*4 + hh;
            int mtf = h >> 9;
            int r   = h & 511;
            int L   = r >> 4;
            int j   = r & 15;
            int m   = mtf*16 + (L & 15);
            int kk  = (L>>4)*8 + (j&7) + ((j>>3)<<4);
            int k   = k0 + kk;
            int ci  = k / 25;
            int rem = k - ci*25;
            int ky  = rem / 5;
            int kx  = rem - ky*5;
            int p = mp0 + m;
            int b = p / PIXIMG;
            int q = p - b*PIXIMG;
            int y = q / WOUT;
            int x = q - y*WOUT;
            As[h] = (_Float16)inp[((b*CIN + ci)*HIN + (y+ky))*WIN + (x+kx)];
        }
        __syncthreads();

        const v16h afrag = *(const v16h*)&As[mt*512 + lane*16];
        const int ktg = k0 >> 5;
#pragma unroll
        for (int ot4 = 0; ot4 < 4; ++ot4) {
            int ot = og*4 + ot4;
            v16h bfrag = Bff[(ktg*16 + ot)*32 + lane];   // packed, L2-resident
            acc[ot4] = __builtin_amdgcn_wmma_f32_16x16x32_f16(
                false, afrag, false, bfrag, (short)0, acc[ot4], false, false);
        }
    }

#pragma unroll
    for (int ot4 = 0; ot4 < 4; ++ot4) {
        int o = (og*4 + ot4)*16 + (lane & 15);
#pragma unroll
        for (int v = 0; v < 8; ++v) {
            int m = mt*16 + v + ((lane>>4)<<3);
            int p = mp0 + m;
            int b = p / PIXIMG;
            int pix = p - b*PIXIMG;
            u0[(b*COUT + o)*PIXIMG + pix] = acc[ot4][v];
        }
    }
}

__global__ void reset_flags_kernel(int* flag, float* gnum, float* gden) {
    *flag = 0; *gnum = 0.f; *gden = 0.f;
}

// u = 0 ; af (f16, fragment layout) = relu(u0). One thread per frag position.
__global__ void init_state_kernel(const float* __restrict__ u0,
                                  float* __restrict__ u,
                                  _Float16* __restrict__ af, int n) {
    int h = blockIdx.x * blockDim.x + threadIdx.x;
    if (h >= n) return;
    int tile = h >> 13;
    int r8   = h & 8191;
    int frag = r8 >> 9;
    int mt = frag >> 3;
    int kt = frag & 7;
    int r  = r8 & 511;
    int L  = r >> 4;
    int j  = r & 15;
    int m  = mt*16 + (L & 15);
    int ci = kt*32 + (L>>4)*8 + (j&7) + ((j>>3)<<4);
    int p  = tile*32 + m;
    int b  = p / PIXIMG;
    int pix = p - b*PIXIMG;
    int idx = (b*COUT + ci)*PIXIMG + pix;
    float v = u0[idx];
    u[idx] = 0.f;
    af[h] = (_Float16)(v > 0.f ? v : 0.f);
}

// ---------------------------------------------------------------------------
// One LCA iteration. Wr (f16, 128 KB) is broadcast into the 320 KB WGP LDS
// with global_load_async_to_lds_b128 (ASYNCcnt). The a-state lives in global
// memory already in f16 A-fragment layout, so each wave's A-operand is a
// contiguous 32 B load per lane. u stays fp32 for the convergence math.
// ---------------------------------------------------------------------------
__global__ __launch_bounds__(256)
void lca_iter_kernel(const float* __restrict__ u0,
                     float*       __restrict__ u,
                     _Float16*    __restrict__ af,
                     const v16h*  __restrict__ Br,
                     const float* __restrict__ th,
                     const int*   __restrict__ flag,
                     float* __restrict__ gnum, float* __restrict__ gden) {
    if (*(volatile const int*)flag) return;     // converged: no-op iteration

    extern __shared__ _Float16 smem[];
    _Float16* sBr = smem;                       // 65536 halves = 128 KB

    const int t = threadIdx.x;
    {   // async copy of the whole packed Wr into LDS (16 B per lane per op)
        const uint4* src = (const uint4*)Br;
        unsigned int ldsbase = (unsigned int)(unsigned long long)sBr;
#pragma unroll
        for (int i = 0; i < 32; ++i) {
            int c = t + 256*i;
            const uint4* g = src + c;
            unsigned int laddr = ldsbase + c*16;
            asm volatile("global_load_async_to_lds_b128 %0, %1, off"
                         :: "v"(laddr), "v"(g) : "memory");
        }
        asm volatile("s_wait_asynccnt 0x0" ::: "memory");
    }
    const int lane = t & 31;
    const int wave = t >> 5;
    const int mt   = wave >> 2;
    const int og   = wave & 3;
    float lnum = 0.f, lden = 0.f;
    __syncthreads();

    const v16h* Af = (const v16h*)af;

    for (int tt = 0; tt < TILES_PER_BLOCK; ++tt) {
        const int tile = blockIdx.x * TILES_PER_BLOCK + tt;
        const int mp0  = tile * 32;

        v8f acc[4] = {};
#pragma unroll
        for (int kt = 0; kt < 8; ++kt) {
            const v16h afrag = Af[(tile*16 + mt*8 + kt)*32 + lane];
#pragma unroll
            for (int ot4 = 0; ot4 < 4; ++ot4) {
                int ot = og*4 + ot4;
                const v16h bfrag = *(const v16h*)&sBr[(kt*16 + ot)*512 + lane*16];
                acc[ot4] = __builtin_amdgcn_wmma_f32_16x16x32_f16(
                    false, afrag, false, bfrag, (short)0, acc[ot4], false, false);
            }
        }
        // all waves must finish reading this tile's a-fragments before any
        // wave rewrites them (no Gauss-Seidel leakage within the block)
        __syncthreads();

        // State update: u' = (1-tau)u + tau(u0 - r); a' = relu(u' - th)
#pragma unroll
        for (int ot4 = 0; ot4 < 4; ++ot4) {
            int o   = (og*4 + ot4)*16 + (lane & 15);
            int kto = o >> 5;
            int kk  = o & 31;
            int Lhi = (kk >> 3) & 1;
            int j   = (kk & 7) + ((kk >> 4) << 3);
            float thv = th[o];
#pragma unroll
            for (int v = 0; v < 8; ++v) {
                int ml = v + ((lane>>4)<<3);          // m within 16-row group
                int p  = mp0 + mt*16 + ml;
                int b  = p / PIXIMG;
                int pix = p - b*PIXIMG;
                int idx = (b*COUT + o)*PIXIMG + pix;
                float uo = u0[idx];
                float uv = u[idx];
                float un = (1.f - TAU)*uv + TAU*(uo - acc[ot4][v]);
                float d  = un - uv;
                lnum += d*d;
                lden += un*un;
                u[idx] = un;
                float an = un - thv;
                af[tile*8192 + (mt*8 + kto)*512 + (Lhi*16 + ml)*16 + j] =
                    (_Float16)(an > 0.f ? an : 0.f);
            }
        }
    }

    for (int off = 16; off > 0; off >>= 1) {
        lnum += __shfl_xor(lnum, off, 32);
        lden += __shfl_xor(lden, off, 32);
    }
    if (lane == 0) { atomicAdd(gnum, lnum); atomicAdd(gden, lden); }
}

__global__ void check_conv_kernel(int* flag, float* gnum, float* gden) {
    if (*flag == 0) {
        float n = *gnum, d = *gden;
        if (!(n > EPS*EPS*d)) *flag = 1;    // ||du|| <= eps*||u||
    }
    *gnum = 0.f; *gden = 0.f;
}

// Exact fp32 output from the fp32 state: a = relu(u - th)
__global__ void export_a_kernel(const float* __restrict__ u,
                                const float* __restrict__ th,
                                float* __restrict__ out, int n) {
    int i = blockIdx.x * blockDim.x + threadIdx.x;
    if (i >= n) return;
    int o = (i / PIXIMG) & (COUT - 1);
    float v = u[i] - th[o];
    out[i] = v > 0.f ? v : 0.f;
}

extern "C" void kernel_launch(void* const* d_in, const int* in_sizes, int n_in,
                              void* d_out, int out_size, void* d_ws, size_t ws_size,
                              hipStream_t stream) {
    const float* inp   = (const float*)d_in[0];   // (16,64,64,64)
    const float* W_ff  = (const float*)d_in[1];   // (256,64,5,5) = [o][k=1600]
    const float* W_rec = (const float*)d_in[2];   // (256,256,1,1) = [o][i]
    const float* th    = (const float*)d_in[3];   // (256,)
    float* out = (float*)d_out;                   // (16,256,60,60)

    char* ws = (char*)d_ws;
    int*   flag = (int*)  (ws + FLAG_OFF);
    float* gnum = (float*)(ws + NUM_OFF);
    float* gden = (float*)(ws + DEN_OFF);
    _Float16* Bff = (_Float16*)(ws + BFF_OFF);
    _Float16* Br  = (_Float16*)(ws + BR_OFF);
    _Float16* af  = (_Float16*)(ws + AF_OFF);
    float* u0 = (float*)(ws + U0_OFF);
    float* u  = (float*)(ws + U_OFF);

    const size_t smem_bytes = 65536 * sizeof(_Float16);   // 128 KB
    (void)hipFuncSetAttribute((const void*)lca_iter_kernel,
                              hipFuncAttributeMaxDynamicSharedMemorySize,
                              (int)smem_bytes);

    pack_b_kernel<<<(KFF*256 + 255)/256, 256, 0, stream>>>(W_ff, Bff, KFF);
    pack_b_kernel<<<(256*256 + 255)/256, 256, 0, stream>>>(W_rec, Br, 256);
    conv_wmma_kernel<<<NPIX/32, 256, 0, stream>>>(inp, (const v16h*)Bff, u0);
    reset_flags_kernel<<<1, 1, 0, stream>>>(flag, gnum, gden);
    init_state_kernel<<<(NELEM + 255)/256, 256, 0, stream>>>(u0, u, af, NELEM);

    for (int it = 0; it < NITER; ++it) {
        lca_iter_kernel<<<ITER_BLOCKS, 256, smem_bytes, stream>>>(
            u0, u, af, (const v16h*)Br, th, flag, gnum, gden);
        check_conv_kernel<<<1, 1, 0, stream>>>(flag, gnum, gden);
    }
    export_a_kernel<<<(NELEM + 255)/256, 256, 0, stream>>>(u, th, out, NELEM);
}
